// SpatioTemporalAttentionGCN_74328704024894
// MI455X (gfx1250) — matrix operand order repaired and checked by reading, not compiled, not run on hardware
//
#include <hip/hip_runtime.h>
#include <hip/hip_bf16.h>
#include <math.h>

// ---------------- problem constants (from reference) ----------------
#define T_STEPS 10
#define NN      50000
#define RR      50000
#define DIN     128
#define DD      64
#define HH      4
#define HDIM    16
#define EE      400000
#define QKV_N   (3*DD)   // 192

typedef __attribute__((ext_vector_type(16))) _Float16 v16h;
typedef __attribute__((ext_vector_type(8)))  _Float16 v8h;
typedef __attribute__((ext_vector_type(8)))  float    v8f;

// ---------------- small utility kernels ----------------
__global__ void k_fill_f32(float* p, float v, long n) {
    long i = (long)blockIdx.x * blockDim.x + threadIdx.x;
    if (i < n) p[i] = v;
}

__global__ void k_f32_to_f16(const float* __restrict__ src, _Float16* __restrict__ dst, long n) {
    long i = (long)blockIdx.x * blockDim.x + threadIdx.x;
    if (i < n) dst[i] = (_Float16)src[i];
}

// ---------------- WMMA GEMM: C[m][n] = sum_k A[m][k]*W[n][k] (+bias[n]) ----------------
// A: MxK f16 row-major, W: NxK f16 row-major (PyTorch-style weight => computes A @ W^T).
// One wave32 per 16x16 output tile. M%16==0, N%16==0, K%32==0.
// A fragment layout (matches B): lane = m + 16*g holds
//   K = g*16 + [0..15] contiguous.
// B fragment layout (32x16): lane = n + 16*g holds
//   K = g*16 + [0..15] contiguous.
__global__ void k_gemm_wmma_f16(const _Float16* __restrict__ A,
                                const _Float16* __restrict__ W,
                                const float*    __restrict__ bias,
                                float*          __restrict__ C,
                                int M, int Ncols, int K, int hasBias)
{
    const int tilesN = Ncols >> 4;
    const int tile   = blockIdx.x;
    const int tm     = tile / tilesN;
    const int tn     = tile - tm * tilesN;
    const int lane   = threadIdx.x & 31;
    const int g      = lane >> 4;      // lane group 0/1
    const int lr     = lane & 15;      // row (A) / col (B) within tile

    const size_t arow = (size_t)(tm * 16 + lr) * (size_t)K;
    const size_t brow = (size_t)(tn * 16 + lr) * (size_t)K;

    v8f c = {};
    for (int k0 = 0; k0 < K; k0 += 32) {
        v8h alo = *(const v8h*)(A + arow + k0 + g * 16);
        v8h ahi = *(const v8h*)(A + arow + k0 + g * 16 + 8);
        v16h a, b;
        b = *(const v16h*)(W + brow + k0 + g * 16);
        #pragma unroll
        for (int i = 0; i < 8; ++i) { a[i] = alo[i]; a[i + 8] = ahi[i]; }
        // 8 args: (neg_a, A, neg_b, B, c_mod, C, reuse_a, reuse_b)
        c = __builtin_amdgcn_wmma_f32_16x16x32_f16(false, a, false, b,
                                                   (short)0, c, false, false);
    }

    // C/D layout: VGPR j, lanes 0-15 -> M=j, lanes 16-31 -> M=j+8; N = lane&15.
    const int col = tn * 16 + lr;
    const float bv = hasBias ? bias[col] : 0.0f;
    #pragma unroll
    for (int j = 0; j < 8; ++j) {
        int row = tm * 16 + j + 8 * g;
        C[(size_t)row * Ncols + col] = c[j] + bv;
    }
}

// ---------------- GCN support kernels ----------------
// deg[dst] += ew  (self-loops handled by deg init = 1.0)
__global__ void k_deg_edges(const int* __restrict__ ei_t, const float* __restrict__ ew_t,
                            float* __restrict__ deg) {
    long e = (long)blockIdx.x * blockDim.x + threadIdx.x;
    if (e < EE) atomicAdd(&deg[ei_t[EE + e]], ew_t[e]);
}

__global__ void k_deg_to_dinv(float* __restrict__ deg) {
    long i = (long)blockIdx.x * blockDim.x + threadIdx.x;
    if (i < NN) { float d = deg[i]; deg[i] = (d > 0.0f) ? rsqrtf(d) : 0.0f; }
}

// agg[n][d] = dinv[n]^2 * hw[n][d]   (self-loop, weight 1)
__global__ void k_agg_self(const float* __restrict__ dinv, const float* __restrict__ hw,
                           float* __restrict__ agg) {
    long i = (long)blockIdx.x * blockDim.x + threadIdx.x;
    if (i < (long)NN * DD) {
        int node = (int)(i >> 6);
        float di = dinv[node];
        agg[i] = di * di * hw[i];
    }
}

// agg[dst][d] += dinv[src]*ew*dinv[dst] * hw[src][d]
__global__ void k_agg_edges(const int* __restrict__ ei_t, const float* __restrict__ ew_t,
                            const float* __restrict__ dinv, const float* __restrict__ hw,
                            float* __restrict__ agg) {
    long tid = (long)blockIdx.x * blockDim.x + threadIdx.x;
    if (tid >= (long)EE * DD) return;
    int e = (int)(tid >> 6);
    int d = (int)(tid & 63);
    int s = ei_t[e];
    int t = ei_t[EE + e];
    float nrm = dinv[s] * ew_t[e] * dinv[t];
    atomicAdd(&agg[(size_t)t * DD + d], nrm * hw[(size_t)s * DD + d]);
}

// h1h[n][d] = f16( relu(agg[n][d] + b[d]) )
__global__ void k_bias_relu_f16(const float* __restrict__ agg, const float* __restrict__ b,
                                _Float16* __restrict__ out) {
    long i = (long)blockIdx.x * blockDim.x + threadIdx.x;
    if (i < (long)NN * DD) {
        int d = (int)(i & 63);
        out[i] = (_Float16)fmaxf(agg[i] + b[d], 0.0f);
    }
}

// scatter layer-2 output into time_embeds (R,T,D), mark presence
__global__ void k_post2_scatter(const float* __restrict__ agg, const float* __restrict__ b,
                                const int* __restrict__ gidx_t, float* __restrict__ te,
                                float* __restrict__ pres, int t) {
    long i = (long)blockIdx.x * blockDim.x + threadIdx.x;
    if (i < (long)NN * DD) {
        int node = (int)(i >> 6);
        int d    = (int)(i & 63);
        int gn   = gidx_t[node];
        float v  = fmaxf(agg[i] + b[d], 0.0f);
        te[((size_t)gn * T_STEPS + t) * DD + d] = v;
        if (d == 0) pres[(size_t)gn * T_STEPS + t] = 1.0f;
    }
}

// ---------------- attention: only row t = T-1 of the score matrix is consumed ----------------
// one thread per (r, head): ao_last[r][h*16 + d] = sum_s softmax_s(q_last . k_s /4 + bias_s) v_s[d]
__global__ void k_attn_last(const float* __restrict__ qkv, const float* __restrict__ pres,
                            float* __restrict__ ao) {
    long idx = (long)blockIdx.x * blockDim.x + threadIdx.x;
    if (idx >= (long)RR * HH) return;
    int r = (int)(idx >> 2);
    int h = (int)(idx & 3);
    const float* base = qkv + (size_t)r * T_STEPS * QKV_N;

    float q[HDIM];
    const float* qp = base + (T_STEPS - 1) * QKV_N + h * HDIM;   // q block = cols [0,64)
    #pragma unroll
    for (int d = 0; d < HDIM; ++d) q[d] = qp[d];

    float sc[T_STEPS];
    float mx = -3.0e38f;
    #pragma unroll
    for (int s = 0; s < T_STEPS; ++s) {
        const float* kp = base + s * QKV_N + DD + h * HDIM;      // k block = cols [64,128)
        float dot = 0.0f;
        #pragma unroll
        for (int d = 0; d < HDIM; ++d) dot += q[d] * kp[d];
        float bias = (pres[(size_t)r * T_STEPS + s] != 0.0f) ? 0.0f : -1.0e9f;
        sc[s] = dot * 0.25f + bias;                              // 1/sqrt(16)
        mx = fmaxf(mx, sc[s]);
    }
    float sum = 0.0f;
    #pragma unroll
    for (int s = 0; s < T_STEPS; ++s) { sc[s] = expf(sc[s] - mx); sum += sc[s]; }
    float inv = 1.0f / sum;

    float acc[HDIM];
    #pragma unroll
    for (int d = 0; d < HDIM; ++d) acc[d] = 0.0f;
    #pragma unroll
    for (int s = 0; s < T_STEPS; ++s) {
        float w = sc[s] * inv;
        const float* vp = base + s * QKV_N + 2 * DD + h * HDIM;  // v block = cols [128,192)
        #pragma unroll
        for (int d = 0; d < HDIM; ++d) acc[d] += w * vp[d];
    }
    float* op = ao + (size_t)r * DD + h * HDIM;
    #pragma unroll
    for (int d = 0; d < HDIM; ++d) op[d] = acc[d];
}

// ---------------- fused layer norms ----------------
// final = LN( w1[T-1]*te[r,T-1] + w2[T-1]*attn_out[r] ; ln1 ); writes f32 + f16
__global__ void k_ln1(const float* __restrict__ te, const float* __restrict__ ao,
                      const float* __restrict__ w1, const float* __restrict__ w2,
                      const float* __restrict__ g, const float* __restrict__ b,
                      float* __restrict__ yf, _Float16* __restrict__ yh) {
    long r = (long)blockIdx.x * blockDim.x + threadIdx.x;
    if (r >= RR) return;
    const float* tp = te + ((size_t)r * T_STEPS + (T_STEPS - 1)) * DD;
    const float* ap = ao + (size_t)r * DD;
    const float* w1p = w1 + (T_STEPS - 1) * DD;
    const float* w2p = w2 + (T_STEPS - 1) * DD;
    float v[DD];
    float mu = 0.0f;
    #pragma unroll
    for (int d = 0; d < DD; ++d) { v[d] = w1p[d] * tp[d] + w2p[d] * ap[d]; mu += v[d]; }
    mu *= (1.0f / DD);
    float var = 0.0f;
    #pragma unroll
    for (int d = 0; d < DD; ++d) { float c = v[d] - mu; var += c * c; }
    var *= (1.0f / DD);
    float rs = rsqrtf(var + 1e-5f);
    #pragma unroll
    for (int d = 0; d < DD; ++d) {
        float y = (v[d] - mu) * rs * g[d] + b[d];
        yf[(size_t)r * DD + d] = y;
        yh[(size_t)r * DD + d] = (_Float16)y;
    }
}

// out = LN( w3*final + w4*fcout ; ln2 ) -> d_out
__global__ void k_ln2(const float* __restrict__ fin, const float* __restrict__ fco,
                      const float* __restrict__ w3, const float* __restrict__ w4,
                      const float* __restrict__ g, const float* __restrict__ b,
                      float* __restrict__ out) {
    long r = (long)blockIdx.x * blockDim.x + threadIdx.x;
    if (r >= RR) return;
    const float* fp = fin + (size_t)r * DD;
    const float* cp = fco + (size_t)r * DD;
    float v[DD];
    float mu = 0.0f;
    #pragma unroll
    for (int d = 0; d < DD; ++d) { v[d] = w3[d] * fp[d] + w4[d] * cp[d]; mu += v[d]; }
    mu *= (1.0f / DD);
    float var = 0.0f;
    #pragma unroll
    for (int d = 0; d < DD; ++d) { float c = v[d] - mu; var += c * c; }
    var *= (1.0f / DD);
    float rs = rsqrtf(var + 1e-5f);
    #pragma unroll
    for (int d = 0; d < DD; ++d)
        out[(size_t)r * DD + d] = (v[d] - mu) * rs * g[d] + b[d];
}

// ---------------- host orchestration ----------------
static inline unsigned nb(long n, int bs) { return (unsigned)((n + bs - 1) / bs); }

extern "C" void kernel_launch(void* const* d_in, const int* in_sizes, int n_in,
                              void* d_out, int out_size, void* d_ws, size_t ws_size,
                              hipStream_t stream) {
    const float* x          = (const float*)d_in[0];   // (T,N,128)
    const int*   edge_index = (const int*)  d_in[1];   // (T,2,E)
    const float* edge_w     = (const float*)d_in[2];   // (T,E)
    const int*   global_idx = (const int*)  d_in[3];   // (T,R)
    const float* gcn1_w     = (const float*)d_in[4];
    const float* gcn1_b     = (const float*)d_in[5];
    const float* gcn2_w     = (const float*)d_in[6];
    const float* gcn2_b     = (const float*)d_in[7];
    const float* inproj_w   = (const float*)d_in[8];
    const float* inproj_b   = (const float*)d_in[9];
    const float* outproj_w  = (const float*)d_in[10];
    const float* outproj_b  = (const float*)d_in[11];
    const float* w1         = (const float*)d_in[12];
    const float* w2         = (const float*)d_in[13];
    const float* w3         = (const float*)d_in[14];
    const float* w4         = (const float*)d_in[15];
    const float* ln1_g      = (const float*)d_in[16];
    const float* ln1_b      = (const float*)d_in[17];
    const float* ln2_g      = (const float*)d_in[18];
    const float* ln2_b      = (const float*)d_in[19];
    const float* fc_w       = (const float*)d_in[20];
    const float* fc_b       = (const float*)d_in[21];
    float* out = (float*)d_out;

    // workspace carve-out (256B aligned)
    char* ws = (char*)d_ws;
    size_t off = 0;
    auto carve = [&](size_t bytes) -> void* {
        void* p = (void*)(ws + off);
        off = (off + bytes + 255) & ~(size_t)255;
        return p;
    };
    _Float16* wg1h  = (_Float16*)carve((size_t)DD * DIN * 2);
    _Float16* wg2h  = (_Float16*)carve((size_t)DD * DD * 2);
    _Float16* wqkvh = (_Float16*)carve((size_t)QKV_N * DD * 2);
    _Float16* wouth = (_Float16*)carve((size_t)DD * DD * 2);
    _Float16* wfch  = (_Float16*)carve((size_t)DD * DD * 2);
    _Float16* xth   = (_Float16*)carve((size_t)NN * DIN * 2);
    float*    hw    = (float*)   carve((size_t)NN * DD * 4);
    _Float16* h1h   = (_Float16*)carve((size_t)NN * DD * 2);
    float*    agg   = (float*)   carve((size_t)NN * DD * 4);
    float*    dinv  = (float*)   carve((size_t)NN * 4);
    float*    te    = (float*)   carve((size_t)RR * T_STEPS * DD * 4);
    _Float16* teh   = (_Float16*)carve((size_t)RR * T_STEPS * DD * 2);
    float*    pres  = (float*)   carve((size_t)RR * T_STEPS * 4);
    float*    qkv   = (float*)   carve((size_t)RR * T_STEPS * QKV_N * 4);
    float*    aoL   = (float*)   carve((size_t)RR * DD * 4);
    _Float16* aoLh  = (_Float16*)carve((size_t)RR * DD * 2);
    float*    atno  = (float*)   carve((size_t)RR * DD * 4);
    float*    finf  = (float*)   carve((size_t)RR * DD * 4);
    _Float16* finh  = (_Float16*)carve((size_t)RR * DD * 2);
    float*    fco   = (float*)   carve((size_t)RR * DD * 4);
    (void)ws_size; (void)in_sizes; (void)n_in; (void)out_size;

    const int BS = 256;

    // weights -> f16
    k_f32_to_f16<<<nb((long)DD * DIN, BS), BS, 0, stream>>>(gcn1_w,    wg1h,  (long)DD * DIN);
    k_f32_to_f16<<<nb((long)DD * DD,  BS), BS, 0, stream>>>(gcn2_w,    wg2h,  (long)DD * DD);
    k_f32_to_f16<<<nb((long)QKV_N*DD, BS), BS, 0, stream>>>(inproj_w,  wqkvh, (long)QKV_N * DD);
    k_f32_to_f16<<<nb((long)DD * DD,  BS), BS, 0, stream>>>(outproj_w, wouth, (long)DD * DD);
    k_f32_to_f16<<<nb((long)DD * DD,  BS), BS, 0, stream>>>(fc_w,      wfch,  (long)DD * DD);

    // zero time_embeds + presence
    k_fill_f32<<<nb((long)RR * T_STEPS * DD, BS), BS, 0, stream>>>(te,   0.0f, (long)RR * T_STEPS * DD);
    k_fill_f32<<<nb((long)RR * T_STEPS,      BS), BS, 0, stream>>>(pres, 0.0f, (long)RR * T_STEPS);

    const int tilesM_N = NN / 16;  // 3125
    for (int t = 0; t < T_STEPS; ++t) {
        const float* x_t  = x          + (size_t)t * NN * DIN;
        const int*   ei_t = edge_index + (size_t)t * 2 * EE;
        const float* ew_t = edge_w     + (size_t)t * EE;
        const int*   gi_t = global_idx + (size_t)t * RR;

        // x_t -> f16
        k_f32_to_f16<<<nb((long)NN * DIN, BS), BS, 0, stream>>>(x_t, xth, (long)NN * DIN);

        // hw = x_t @ gcn1_w^T   (no bias: bias applied after aggregation)
        k_gemm_wmma_f16<<<tilesM_N * (DD / 16), 32, 0, stream>>>(xth, wg1h, nullptr, hw,
                                                                 NN, DD, DIN, 0);
        // degrees: self-loop weight 1, then edge weights into dst; then rsqrt
        k_fill_f32<<<nb(NN, BS), BS, 0, stream>>>(dinv, 1.0f, NN);
        k_deg_edges<<<nb(EE, BS), BS, 0, stream>>>(ei_t, ew_t, dinv);
        k_deg_to_dinv<<<nb(NN, BS), BS, 0, stream>>>(dinv);

        // layer 1 aggregation
        k_agg_self<<<nb((long)NN * DD, BS), BS, 0, stream>>>(dinv, hw, agg);
        k_agg_edges<<<nb((long)EE * DD, BS), BS, 0, stream>>>(ei_t, ew_t, dinv, hw, agg);
        k_bias_relu_f16<<<nb((long)NN * DD, BS), BS, 0, stream>>>(agg, gcn1_b, h1h);

        // layer 2: hw = h1 @ gcn2_w^T, aggregate, scatter into te
        k_gemm_wmma_f16<<<tilesM_N * (DD / 16), 32, 0, stream>>>(h1h, wg2h, nullptr, hw,
                                                                 NN, DD, DD, 0);
        k_agg_self<<<nb((long)NN * DD, BS), BS, 0, stream>>>(dinv, hw, agg);
        k_agg_edges<<<nb((long)EE * DD, BS), BS, 0, stream>>>(ei_t, ew_t, dinv, hw, agg);
        k_post2_scatter<<<nb((long)NN * DD, BS), BS, 0, stream>>>(agg, gcn2_b, gi_t, te, pres, t);
    }

    // te -> f16, qkv = te @ in_proj^T + b   (M = R*T = 500000)
    k_f32_to_f16<<<nb((long)RR * T_STEPS * DD, BS), BS, 0, stream>>>(te, teh,
                                                                     (long)RR * T_STEPS * DD);
    k_gemm_wmma_f16<<<(RR * T_STEPS / 16) * (QKV_N / 16), 32, 0, stream>>>(
        teh, wqkvh, inproj_b, qkv, RR * T_STEPS, QKV_N, DD, 1);

    // attention (last time step only feeds the output)
    k_attn_last<<<nb((long)RR * HH, BS), BS, 0, stream>>>(qkv, pres, aoL);

    // attn_out = aoL @ out_proj^T + b
    k_f32_to_f16<<<nb((long)RR * DD, BS), BS, 0, stream>>>(aoL, aoLh, (long)RR * DD);
    k_gemm_wmma_f16<<<(RR / 16) * (DD / 16), 32, 0, stream>>>(aoLh, wouth, outproj_b, atno,
                                                              RR, DD, DD, 1);

    // final = LN1(w1*te_last + w2*attn_out)
    k_ln1<<<nb(RR, BS), BS, 0, stream>>>(te, atno, w1, w2, ln1_g, ln1_b, finf, finh);

    // fcout = final @ fc_w^T + b
    k_gemm_wmma_f16<<<(RR / 16) * (DD / 16), 32, 0, stream>>>(finh, wfch, fc_b, fco,
                                                              RR, DD, DD, 1);

    // d_out = LN2(w3*final + w4*fcout)
    k_ln2<<<nb(RR, BS), BS, 0, stream>>>(finf, fco, w3, w4, ln2_g, ln2_b, out);
}
